// TensorProductConvLayer_78271484002959
// MI455X (gfx1250) — matrix operand order repaired and checked by reading, not compiled, not run on hardware
//
#include <hip/hip_runtime.h>

// ---------------- problem constants ----------------
constexpr int kNS = 16, kNV = 8;
constexpr int kNodes = 20000;
constexpr int kEdges = 320000;
constexpr int kFeat  = 64;            // edge feature / hidden dim
constexpr int kD     = kNS + 3 * kNV; // 40
constexpr int kWN    = 576;           // W_NUMEL
constexpr float kEps = 1e-5f;
constexpr float kA0  = 0.20412414523193154f;  // 1/sqrt(NS+NV)
constexpr float kRS3 = 0.57735026918962584f;  // 1/sqrt(3)

// ---------------- WMMA types ----------------
typedef __attribute__((ext_vector_type(16))) __bf16 v16bf;
typedef __attribute__((ext_vector_type(8)))  float  v8f;

union BF16x16 {
  v16bf v;
  __bf16 b[16];
  unsigned short us[16];
  uint4 q[2];
};

__device__ __forceinline__ unsigned short f2bf(float f) {
#if defined(__AMDGCN__)
  // native bf16 convert on gfx1250 (single v_cvt, RNE)
  __bf16 b = (__bf16)f;
  return __builtin_bit_cast(unsigned short, b);
#else
  unsigned u = __float_as_uint(f);
  u += 0x7FFFu + ((u >> 16) & 1u);
  return (unsigned short)(u >> 16);
#endif
}

__device__ __forceinline__ v8f wmma_bf16(const BF16x16& a, const BF16x16& b, v8f c) {
  return __builtin_amdgcn_wmma_f32_16x16x32_bf16(false, a.v, false, b.v,
                                                 (short)0, c, false, false);
}

// B-frag ([outC][k] layout), element i => K = kb + (i&7) + ((i>>3)<<4) + half*8
__device__ __forceinline__ BF16x16 ldB(const unsigned short* p) {
  BF16x16 r;
  p -= ((threadIdx.x >> 4) & 1) * 8;
  r.q[0] = *reinterpret_cast<const uint4*>(p);
  r.q[1] = *reinterpret_cast<const uint4*>(p + 16);
  return r;
}

// Wave-local LDS ordering: DS ops are in-order within a wave; waiting DScnt==0
// guarantees prior stores are visible to subsequent cross-lane loads.
#define WAVE_LDS_SYNC() asm volatile("s_wait_dscnt 0" ::: "memory")

// ---------------- per-wave LDS scratch ----------------
struct WaveScratch {
  unsigned short h[16][72];  // bf16 hidden acts, padded rows (144B stride)
  float xs_s[16][16];        // xs * shs
  float xs_r[16][16];        // xs
  float xv[16][24];          // vector feats (u-major, m-minor)
  float dotv[16][8];         // (xv . shv)/sqrt(3)
  float shv[16][3];
  float shs[16];
  int   dsti[16];
};

// ---------------- kernels ----------------
__global__ void k_zero(float* p, int n) {
  int i = blockIdx.x * 256 + threadIdx.x;
  if (i < n) p[i] = 0.f;
}

__global__ void k_prep(const float* __restrict__ fc1_w, const float* __restrict__ fc2_w,
                       unsigned short* __restrict__ w1t, unsigned short* __restrict__ w2t) {
  int i = blockIdx.x * 256 + threadIdx.x;
  if (i < kFeat * kFeat) {                       // fc1_w[k][c] -> w1t[c][k]
    int k = i >> 6, c = i & 63;
    w1t[c * 64 + k] = f2bf(fc1_w[i]);
  } else if (i < kFeat * kFeat + kFeat * kWN) {  // fc2_w[k][c] -> w2t[c][k]
    int j = i - kFeat * kFeat;
    int k = j / kWN, c = j % kWN;
    w2t[c * 64 + k] = f2bf(fc2_w[j]);
  }
}

__global__ __launch_bounds__(256) void k_main(
    const float* __restrict__ node_attr,
    const float* __restrict__ edge_attr,
    const float* __restrict__ edge_sh,
    const float* __restrict__ fc1_b,
    const float* __restrict__ fc2_b,
    const int*   __restrict__ eidx,
    const unsigned short* __restrict__ w1t,  // [64][64] bf16
    const unsigned short* __restrict__ w2t,  // [576][64] bf16
    float* __restrict__ agg)
{
  __shared__ WaveScratch wsAll[8];
  const int tid  = threadIdx.x;
  const int wv   = tid >> 5;
  const int lane = tid & 31;
  const int lh   = lane >> 4;   // lane half
  const int eL   = lane & 15;
  WaveScratch& ws = wsAll[wv];
  const int waveGlobal = blockIdx.x * 8 + wv;

  for (int g = 0; g < 4; ++g) {
    const int ebase = (waveGlobal * 4 + g) * 16;
    if (g + 1 < 4)
      __builtin_prefetch(edge_attr + (size_t)(ebase + 16 + eL) * kFeat, 0, 1);

    // -------- Phase 1: per-edge gather tables --------
    {
      const int e   = ebase + eL;
      const int src = eidx[e];
      if (lh == 0) {
        const float sh0 = edge_sh[e * 4 + 0];
        ws.shs[eL]  = sh0;
        ws.dsti[eL] = eidx[kEdges + e];
        const float* xp = node_attr + (size_t)src * kD;
        #pragma unroll
        for (int u = 0; u < kNS; ++u) {
          float x = xp[u];
          ws.xs_r[eL][u] = x;
          ws.xs_s[eL][u] = x * sh0;
        }
      } else {
        const float s0 = edge_sh[e * 4 + 1];
        const float s1 = edge_sh[e * 4 + 2];
        const float s2 = edge_sh[e * 4 + 3];
        ws.shv[eL][0] = s0; ws.shv[eL][1] = s1; ws.shv[eL][2] = s2;
        const float* xp = node_attr + (size_t)src * kD + kNS;
        #pragma unroll
        for (int u = 0; u < kNV; ++u) {
          float x0 = xp[u*3+0], x1 = xp[u*3+1], x2 = xp[u*3+2];
          ws.xv[eL][u*3+0] = x0; ws.xv[eL][u*3+1] = x1; ws.xv[eL][u*3+2] = x2;
          ws.dotv[eL][u] = (x0*s0 + x1*s1 + x2*s2) * kRS3;
        }
      }
    }
    WAVE_LDS_SYNC();

    // -------- Phase 2: GEMM1  h = relu(edge_attr @ fc1_w + b) --------
    // A-frag (16x32 bf16): lane row M=eL; element i -> K = kb + (i&7) + ((i>>3)<<4) + lh*8
    BF16x16 A0f, A1f;
    {
      const float* pa = edge_attr + (size_t)(ebase + eL) * kFeat + lh * 8;
      float4 r0 = *(const float4*)(pa +  0), r1 = *(const float4*)(pa +  4);
      float4 r2 = *(const float4*)(pa + 16), r3 = *(const float4*)(pa + 20);
      float4 r4 = *(const float4*)(pa + 32), r5 = *(const float4*)(pa + 36);
      float4 r6 = *(const float4*)(pa + 48), r7 = *(const float4*)(pa + 52);
      A0f.us[0]=f2bf(r0.x); A0f.us[1]=f2bf(r0.y); A0f.us[2]=f2bf(r0.z); A0f.us[3]=f2bf(r0.w);
      A0f.us[4]=f2bf(r1.x); A0f.us[5]=f2bf(r1.y); A0f.us[6]=f2bf(r1.z); A0f.us[7]=f2bf(r1.w);
      A0f.us[8]=f2bf(r2.x); A0f.us[9]=f2bf(r2.y); A0f.us[10]=f2bf(r2.z); A0f.us[11]=f2bf(r2.w);
      A0f.us[12]=f2bf(r3.x); A0f.us[13]=f2bf(r3.y); A0f.us[14]=f2bf(r3.z); A0f.us[15]=f2bf(r3.w);
      A1f.us[0]=f2bf(r4.x); A1f.us[1]=f2bf(r4.y); A1f.us[2]=f2bf(r4.z); A1f.us[3]=f2bf(r4.w);
      A1f.us[4]=f2bf(r5.x); A1f.us[5]=f2bf(r5.y); A1f.us[6]=f2bf(r5.z); A1f.us[7]=f2bf(r5.w);
      A1f.us[8]=f2bf(r6.x); A1f.us[9]=f2bf(r6.y); A1f.us[10]=f2bf(r6.z); A1f.us[11]=f2bf(r6.w);
      A1f.us[12]=f2bf(r7.x); A1f.us[13]=f2bf(r7.y); A1f.us[14]=f2bf(r7.z); A1f.us[15]=f2bf(r7.w);
    }
    #pragma unroll
    for (int nt = 0; nt < 4; ++nt) {
      const int c = nt * 16 + eL;
      BF16x16 b0 = ldB(w1t + c * 64 + 0  + lh * 16);
      BF16x16 b1 = ldB(w1t + c * 64 + 32 + lh * 16);
      const float bias = fc1_b[c];
      v8f acc;
      #pragma unroll
      for (int r = 0; r < 8; ++r) acc[r] = bias;
      acc = wmma_bf16(A0f, b0, acc);
      acc = wmma_bf16(A1f, b1, acc);
      #pragma unroll
      for (int r = 0; r < 8; ++r)
        ws.h[r + 8 * lh][c] = f2bf(fmaxf(acc[r], 0.f));  // relu + bf16 restage
    }
    WAVE_LDS_SYNC();

    // -------- Phase 3: GEMM2 tiles consumed in-register --------
    BF16x16 H0, H1;
    {
      const unsigned short* hp = &ws.h[eL][0];
      H0.q[0] = *reinterpret_cast<const uint4*>(hp +  0 + lh * 8);
      H0.q[1] = *reinterpret_cast<const uint4*>(hp + 16 + lh * 8);
      H1.q[0] = *reinterpret_cast<const uint4*>(hp + 32 + lh * 8);
      H1.q[1] = *reinterpret_cast<const uint4*>(hp + 48 + lh * 8);
    }
    v8f out0;
    float o1[3][8];
    #pragma unroll
    for (int r = 0; r < 8; ++r) { out0[r] = 0.f; o1[0][r] = 0.f; o1[1][r] = 0.f; o1[2][r] = 0.f; }

    // tiles 0..15: w1[u=t][w'=col]  -> out0 += D * (xs*shs)[e][t]
    #pragma unroll 2
    for (int t = 0; t < 16; ++t) {
      const int c = t * 16 + eL;
      BF16x16 b0 = ldB(w2t + c * 64 + 0  + lh * 16);
      BF16x16 b1 = ldB(w2t + c * 64 + 32 + lh * 16);
      const float bias = fc2_b[c];
      v8f d;
      #pragma unroll
      for (int r = 0; r < 8; ++r) d[r] = bias;
      d = wmma_bf16(H0, b0, d);
      d = wmma_bf16(H1, b1, d);
      #pragma unroll
      for (int r = 0; r < 8; ++r) out0[r] += d[r] * ws.xs_s[r + 8 * lh][t];
    }
    // tiles 16..23: w2[u][v] -> out1[v][m] += D * xs[e][u] * shv[e][m]
    #pragma unroll 2
    for (int tt = 0; tt < 8; ++tt) {
      const int c = (16 + tt) * 16 + eL;
      BF16x16 b0 = ldB(w2t + c * 64 + 0  + lh * 16);
      BF16x16 b1 = ldB(w2t + c * 64 + 32 + lh * 16);
      const float bias = fc2_b[c];
      v8f d;
      #pragma unroll
      for (int r = 0; r < 8; ++r) d[r] = bias;
      d = wmma_bf16(H0, b0, d);
      d = wmma_bf16(H1, b1, d);
      const int u = 2 * tt + (eL >> 3);
      #pragma unroll
      for (int r = 0; r < 8; ++r) {
        const int er = r + 8 * lh;
        const float tmp = d[r] * ws.xs_r[er][u];
        o1[0][r] += tmp * ws.shv[er][0];
        o1[1][r] += tmp * ws.shv[er][1];
        o1[2][r] += tmp * ws.shv[er][2];
      }
    }
    // tiles 24..27: w3[u][v] -> out1[v][m] += D * shs[e] * xv[e][u][m]
    #pragma unroll
    for (int tt = 0; tt < 4; ++tt) {
      const int c = (24 + tt) * 16 + eL;
      BF16x16 b0 = ldB(w2t + c * 64 + 0  + lh * 16);
      BF16x16 b1 = ldB(w2t + c * 64 + 32 + lh * 16);
      const float bias = fc2_b[c];
      v8f d;
      #pragma unroll
      for (int r = 0; r < 8; ++r) d[r] = bias;
      d = wmma_bf16(H0, b0, d);
      d = wmma_bf16(H1, b1, d);
      const int u = 2 * tt + (eL >> 3);
      #pragma unroll
      for (int r = 0; r < 8; ++r) {
        const int er = r + 8 * lh;
        const float tmp = d[r] * ws.shs[er];
        o1[0][r] += tmp * ws.xv[er][u * 3 + 0];
        o1[1][r] += tmp * ws.xv[er][u * 3 + 1];
        o1[2][r] += tmp * ws.xv[er][u * 3 + 2];
      }
    }
    // tiles 28..35: w4[u][w'] -> out0 += D * dot[e][u]
    #pragma unroll 2
    for (int tt = 0; tt < 8; ++tt) {
      const int c = (28 + tt) * 16 + eL;
      BF16x16 b0 = ldB(w2t + c * 64 + 0  + lh * 16);
      BF16x16 b1 = ldB(w2t + c * 64 + 32 + lh * 16);
      const float bias = fc2_b[c];
      v8f d;
      #pragma unroll
      for (int r = 0; r < 8; ++r) d[r] = bias;
      d = wmma_bf16(H0, b0, d);
      d = wmma_bf16(H1, b1, d);
      #pragma unroll
      for (int r = 0; r < 8; ++r) out0[r] += d[r] * ws.dotv[r + 8 * lh][tt];
    }

    // -------- Phase 4: scatter-add to agg (both u-halves of out1 sum here) --------
    const int vch = eL & 7;
    #pragma unroll
    for (int r = 0; r < 8; ++r) {
      const int er  = r + 8 * lh;
      const int dst = ws.dsti[er];
      float* ap = agg + (size_t)dst * kD;
      atomicAdd(ap + eL,                 out0[r]  * kA0);
      atomicAdd(ap + kNS + vch * 3 + 0,  o1[0][r] * kA0);
      atomicAdd(ap + kNS + vch * 3 + 1,  o1[1][r] * kA0);
      atomicAdd(ap + kNS + vch * 3 + 2,  o1[2][r] * kA0);
    }
    WAVE_LDS_SYNC();   // tables safe to overwrite next group
  }
}

__global__ __launch_bounds__(256) void k_stats(const float* __restrict__ agg,
                                               const float* __restrict__ na,
                                               float* __restrict__ stats) {
  __shared__ float s[40];
  const int tid = threadIdx.x;
  if (tid < 40) s[tid] = 0.f;
  __syncthreads();
  const int i = blockIdx.x * 256 + tid;
  float contrib[40];
  if (i < kNodes) {
    float v[40];
    #pragma unroll
    for (int c = 0; c < 40; ++c) v[c] = agg[i * kD + c] + na[i * kD + c];
    #pragma unroll
    for (int c = 0; c < 16; ++c) { contrib[c] = v[c]; contrib[16 + c] = v[c] * v[c]; }
    #pragma unroll
    for (int u = 0; u < 8; ++u) {
      float a = v[16 + u * 3], b = v[17 + u * 3], cc = v[18 + u * 3];
      contrib[32 + u] = a * a + b * b + cc * cc;
    }
  } else {
    #pragma unroll
    for (int c = 0; c < 40; ++c) contrib[c] = 0.f;
  }
  const int lane = tid & 31;
  #pragma unroll
  for (int j = 0; j < 40; ++j) {
    float x = contrib[j];
    #pragma unroll
    for (int o = 16; o > 0; o >>= 1) x += __shfl_down(x, o, 32);
    if (lane == 0) atomicAdd(&s[j], x);
  }
  __syncthreads();
  if (tid < 40) atomicAdd(&stats[tid], s[tid]);
}

__global__ void k_norm(const float* __restrict__ agg, const float* __restrict__ na,
                       const float* __restrict__ stats,
                       const float* __restrict__ gs, const float* __restrict__ bs,
                       const float* __restrict__ gv, float* __restrict__ out) {
  const int idx = blockIdx.x * 256 + threadIdx.x;
  if (idx >= kNodes * kD) return;
  const int c = idx % kD;
  const float v = agg[idx] + na[idx];
  float o;
  if (c < kNS) {
    const float mu  = stats[c]      * (1.f / kNodes);
    const float var = stats[16 + c] * (1.f / kNodes) - mu * mu;
    o = (v - mu) * rsqrtf(var + kEps) * gs[c] + bs[c];
  } else {
    const int vc = (c - kNS) / 3;
    const float vn = stats[32 + vc] * (1.f / (kNodes * 3));
    o = v * rsqrtf(vn + kEps) * gv[vc];
  }
  out[idx] = o;
}

// ---------------- host launcher ----------------
extern "C" void kernel_launch(void* const* d_in, const int* in_sizes, int n_in,
                              void* d_out, int out_size, void* d_ws, size_t ws_size,
                              hipStream_t stream) {
  const float* node_attr = (const float*)d_in[0];
  const float* edge_attr = (const float*)d_in[1];
  const float* edge_sh   = (const float*)d_in[2];
  const float* fc1_w     = (const float*)d_in[3];
  const float* fc1_b     = (const float*)d_in[4];
  const float* fc2_w     = (const float*)d_in[5];
  const float* fc2_b     = (const float*)d_in[6];
  const float* gs        = (const float*)d_in[7];
  const float* bs        = (const float*)d_in[8];
  const float* gv        = (const float*)d_in[9];
  const int*   eidx      = (const int*)d_in[10];

  // d_ws layout: agg[800000] f32 | stats[40] f32 | w1t[4096] bf16 | w2t[36864] bf16
  float* agg   = (float*)d_ws;
  float* stats = agg + kNodes * kD;
  unsigned short* w1t = (unsigned short*)((char*)d_ws + (size_t)(kNodes * kD + 40) * 4);
  unsigned short* w2t = w1t + kFeat * kFeat;

  const int nZero = kNodes * kD + 40;
  k_zero<<<(nZero + 255) / 256, 256, 0, stream>>>(agg, nZero);

  const int nPrep = kFeat * kFeat + kFeat * kWN;
  k_prep<<<(nPrep + 255) / 256, 256, 0, stream>>>(fc1_w, fc2_w, w1t, w2t);

  // 625 blocks * 8 waves * 4 groups * 16 edges = 320000 edges
  k_main<<<625, 256, 0, stream>>>(node_attr, edge_attr, edge_sh, fc1_b, fc2_b,
                                  eidx, w1t, w2t, agg);

  k_stats<<<(kNodes + 255) / 256, 256, 0, stream>>>(agg, node_attr, stats);

  k_norm<<<(kNodes * kD + 255) / 256, 256, 0, stream>>>(agg, node_attr, stats,
                                                        gs, bs, gv, (float*)d_out);
}
